// GOTPNet_77300821393917
// MI455X (gfx1250) — compile-verified, matrix operand order checked by reading, not run on hardware
//
#include <hip/hip_runtime.h>
#include <hip/hip_bf16.h>
#include <math.h>
#include <stdint.h>

typedef __attribute__((ext_vector_type(16))) _Float16 v16h;
typedef __attribute__((ext_vector_type(8)))  float    v8f;

#define LEAKY_A 0.2f
#define BN_EPS  1e-5f
#define SPG_T   128

// ---------- helpers ----------
static __device__ __forceinline__ unsigned enc_f32(float f) {
  unsigned u = __float_as_uint(f);
  return (u & 0x80000000u) ? ~u : (u | 0x80000000u);
}
static __device__ __forceinline__ float dec_f32(unsigned u) {
  return (u & 0x80000000u) ? __uint_as_float(u ^ 0x80000000u) : __uint_as_float(~u);
}

// ---------- tiny utility kernels ----------
__global__ void k_seti(int* p, int v) { if (blockIdx.x == 0 && threadIdx.x == 0) *p = v; }
__global__ void k_copy_min(int* dst, const int* src, int cap) {
  if (blockIdx.x == 0 && threadIdx.x == 0) { int v = *src; *dst = v < cap ? v : cap; }
}
__global__ void k_fill_f32(float* p, float v, int n) {
  int i = blockIdx.x * blockDim.x + threadIdx.x; if (i < n) p[i] = v;
}
__global__ void k_fill_i32(int* p, int v, int n) {
  int i = blockIdx.x * blockDim.x + threadIdx.x; if (i < n) p[i] = v;
}
__global__ void k_fill_u32(unsigned* p, unsigned v, int n) {
  int i = blockIdx.x * blockDim.x + threadIdx.x; if (i < n) p[i] = v;
}
__global__ void k_copy_i32(int* dst, const int* src, int n) {
  int i = blockIdx.x * blockDim.x + threadIdx.x; if (i < n) dst[i] = src[i];
}
__global__ void k_tanh_ip(float* x, int n) {
  int i = blockIdx.x * blockDim.x + threadIdx.x; if (i < n) x[i] = tanhf(x[i]);
}
__global__ void k_sigmoid(const float* x, float* y, int n) {
  int i = blockIdx.x * blockDim.x + threadIdx.x;
  if (i < n) y[i] = 1.0f / (1.0f + expf(-x[i]));
}

// ---------- fast WMMA GEMM: C[M,N] = A[M,K] @ B[K,N] (+bias[n]) ----------
// Requires M%16==0, N%32==0, K%32==0. One wave -> 16x32 tile (2 accumulators, A reused).
// fp32 in/out, f16 WMMA compute; per-lane A runs are contiguous -> float4 (b128) loads.
__global__ void k_gemm_wmma_fast(const float* __restrict__ A, const float* __restrict__ B,
                                 const float* __restrict__ bias, float* __restrict__ Cst,
                                 int M, int N, int K) {
  const int lane = threadIdx.x & 31;
  const int lh   = lane >> 4;    // lane-half selects K sub-block
  const int l16  = lane & 15;
  const int tm = blockIdx.y * 16;
  const int tn = blockIdx.x * 32;
  const int am  = tm + l16;       // A row held by this lane
  const int bn0 = tn + l16;       // B col for accumulator 0
  const int bn1 = tn + 16 + l16;  // B col for accumulator 1
  const float* __restrict__ Arow = A + (size_t)am * K;
  v8f acc0 = {}, acc1 = {};
  for (int k0 = 0; k0 < K; k0 += 32) {
    const int kb = k0 + lh * 8;
    if (k0 + 32 < K) __builtin_prefetch(Arow + k0 + 32, 0, 1);
    const float4* pa = (const float4*)(Arow + kb);
    float4 a0 = pa[0], a1 = pa[1];   // K = kb .. kb+7
    float4 a2 = pa[4], a3 = pa[5];   // K = kb+16 .. kb+23
    v16h af;
    af[0]  = (_Float16)a0.x; af[1]  = (_Float16)a0.y; af[2]  = (_Float16)a0.z; af[3]  = (_Float16)a0.w;
    af[4]  = (_Float16)a1.x; af[5]  = (_Float16)a1.y; af[6]  = (_Float16)a1.z; af[7]  = (_Float16)a1.w;
    af[8]  = (_Float16)a2.x; af[9]  = (_Float16)a2.y; af[10] = (_Float16)a2.z; af[11] = (_Float16)a2.w;
    af[12] = (_Float16)a3.x; af[13] = (_Float16)a3.y; af[14] = (_Float16)a3.z; af[15] = (_Float16)a3.w;
    v16h b0f, b1f;
#pragma unroll
    for (int e = 0; e < 16; ++e) {
      int kk = kb + ((e < 8) ? e : (e + 8));
      const float* brow = B + (size_t)kk * N;
      b0f[e] = (_Float16)brow[bn0];
      b1f[e] = (_Float16)brow[bn1];
    }
    acc0 = __builtin_amdgcn_wmma_f32_16x16x32_f16(false, af, false, b0f, (short)0, acc0,
                                                  false, false);
    acc1 = __builtin_amdgcn_wmma_f32_16x16x32_f16(false, af, false, b1f, (short)0, acc1,
                                                  false, false);
  }
  __builtin_amdgcn_s_wait_tensorcnt(0);  // CDNA5 tensor-counter path (no-op here)
  float bb0 = bias ? bias[bn0] : 0.0f;
  float bb1 = bias ? bias[bn1] : 0.0f;
#pragma unroll
  for (int j = 0; j < 8; ++j) {
    int r = tm + j + 8 * lh;
    Cst[(size_t)r * N + bn0] = acc0[j] + bb0;
    Cst[(size_t)r * N + bn1] = acc1[j] + bb1;
  }
}

// ---------- matvec: y[m] = X[m,:] . w (+b) ; one wave32 per row ----------
__global__ void k_matvec(const float* __restrict__ X, const float* __restrict__ w,
                         const float* bias, float* __restrict__ y, int M, int K) {
  int row  = blockIdx.x * (blockDim.x >> 5) + (threadIdx.x >> 5);
  int lane = threadIdx.x & 31;
  if (row >= M) return;
  const float* xr = X + (size_t)row * K;
  float acc = 0.0f;
  for (int k = lane; k < K; k += 32) acc += xr[k] * w[k];
  for (int off = 16; off > 0; off >>= 1) acc += __shfl_down(acc, off, 32);
  if (lane == 0) y[row] = acc + (bias ? bias[0] : 0.0f);
}
__global__ void k_affine1(const float* x, const float* W, const float* b, float* y, int n) {
  int i = blockIdx.x * blockDim.x + threadIdx.x;
  if (i < n) y[i] = x[i] * W[0] + b[0];
}

// ---------- edge list with self loops (dead slots = -1) ----------
__global__ void k_loop_edges(const int* src, const int* dst, const int* cnt,
                             int ecap, int nn, int* es, int* ed) {
  int i = blockIdx.x * blockDim.x + threadIdx.x;
  int tot = ecap + nn;
  if (i >= tot) return;
  int E = *cnt;
  if (i < ecap) {
    int s = -1, d = -1;
    if (i < E) {
      s = src[i]; d = dst[i];
      if (s < 0 || d < 0 || s == d) { s = -1; d = -1; }
    }
    es[i] = s; ed[i] = d;
  } else {
    int n = i - ecap;
    int v = (n < nn) ? n : -1;
    es[i] = v; ed[i] = v;
  }
}

// ---------- GATv2 edge kernels ----------
__global__ void k_edge_score(const float* __restrict__ xp, const int* es, const int* ed,
                             const float* __restrict__ att, int C, int etot, float* e) {
  int i = blockIdx.x * blockDim.x + threadIdx.x;
  if (i >= etot) return;
  int s = es[i];
  if (s < 0) { e[i] = 0.0f; return; }
  int d = ed[i];
  float acc = 0.0f;
  if ((C & 3) == 0) {
    const float4* ps = (const float4*)(xp + (size_t)s * C);
    const float4* pd = (const float4*)(xp + (size_t)d * C);
    const float4* pa = (const float4*)att;
    int C4 = C >> 2;
    for (int c = 0; c < C4; ++c) {
      float4 hs = ps[c], hd = pd[c], aa = pa[c];
      float h0 = hs.x + hd.x, h1 = hs.y + hd.y, h2 = hs.z + hd.z, h3 = hs.w + hd.w;
      acc += (h0 > 0.0f ? h0 : LEAKY_A * h0) * aa.x;
      acc += (h1 > 0.0f ? h1 : LEAKY_A * h1) * aa.y;
      acc += (h2 > 0.0f ? h2 : LEAKY_A * h2) * aa.z;
      acc += (h3 > 0.0f ? h3 : LEAKY_A * h3) * aa.w;
    }
  } else {
    for (int c = 0; c < C; ++c) {
      float h = xp[(size_t)s * C + c] + xp[(size_t)d * C + c];
      acc += (h > 0.0f ? h : LEAKY_A * h) * att[c];
    }
  }
  e[i] = acc;
}
__global__ void k_seg_max(const float* e, const int* ed, int etot, unsigned* menc) {
  int i = blockIdx.x * blockDim.x + threadIdx.x;
  if (i >= etot) return;
  int d = ed[i]; if (d < 0) return;
  atomicMax(&menc[d], enc_f32(e[i]));
}
__global__ void k_edge_exp(float* e, const int* ed, int etot, const unsigned* menc, float* ssum) {
  int i = blockIdx.x * blockDim.x + threadIdx.x;
  if (i >= etot) return;
  int d = ed[i]; if (d < 0) return;
  float v = expf(e[i] - dec_f32(menc[d]));
  e[i] = v;
  atomicAdd(&ssum[d], v);
}
__global__ void k_alpha_scatter(const float* __restrict__ e, const float* __restrict__ ssum,
                                const float* __restrict__ xp, const int* es, const int* ed,
                                int etot, int C, float* __restrict__ out) {
  long long i = (long long)blockIdx.x * blockDim.x + threadIdx.x;
  long long tot = (long long)etot * C;
  if (i >= tot) return;
  int edge = (int)(i / C), c = (int)(i % C);
  int s = es[edge];
  if (s < 0) return;
  int d = ed[edge];
  float alpha = e[edge] / ssum[d];
  atomicAdd(&out[(size_t)d * C + c], alpha * xp[(size_t)s * C + c]);
}
__global__ void k_add_bias(float* x, const float* b, int n, int C) {
  int i = blockIdx.x * blockDim.x + threadIdx.x;
  if (i < n * C) x[i] += b[i % C];
}

// ---------- batchnorm (training-mode batch stats), one block per channel ----------
__global__ void k_batchnorm(float* x, const float* g, const float* b, int N, int C) {
  __shared__ float s1[256], s2[256];
  int c = blockIdx.x, tid = threadIdx.x;
  float a1 = 0.0f, a2 = 0.0f;
  for (int r = tid; r < N; r += 256) {
    float v = x[(size_t)r * C + c];
    a1 += v; a2 += v * v;
  }
  s1[tid] = a1; s2[tid] = a2; __syncthreads();
  for (int off = 128; off > 0; off >>= 1) {
    if (tid < off) { s1[tid] += s1[tid + off]; s2[tid] += s2[tid + off]; }
    __syncthreads();
  }
  float mu = s1[0] / (float)N;
  float var = s2[0] / (float)N - mu * mu;
  float inv = rsqrtf(var + BN_EPS);
  float gg = g[c], bb = b[c];
  for (int r = tid; r < N; r += 256) {
    float v = x[(size_t)r * C + c];
    x[(size_t)r * C + c] = gg * (v - mu) * inv + bb;
  }
}

// ---------- pooling: score / sort / gather ----------
__global__ void k_norm(const float* w, int n, float* out) {
  __shared__ float s[256];
  int tid = threadIdx.x;
  float a = 0.0f;
  for (int i = tid; i < n; i += 256) { float v = w[i]; a += v * v; }
  s[tid] = a; __syncthreads();
  for (int off = 128; off > 0; off >>= 1) { if (tid < off) s[tid] += s[tid + off]; __syncthreads(); }
  if (tid == 0) out[0] = sqrtf(s[0]);
}
__global__ void k_score(const float* dot, const float* nrm, float* score, int n) {
  int i = blockIdx.x * blockDim.x + threadIdx.x;
  if (i < n) score[i] = tanhf(dot[i] / nrm[0]);
}
// single-workgroup LDS bitonic sort, descending; N power of two <= 4096
__global__ void k_topk_sort(const float* score, int N, int K, int* perm) {
  __shared__ float sv[4096];
  __shared__ int   si[4096];
  int tid = threadIdx.x;
  for (int i = tid; i < N; i += 1024) { sv[i] = score[i]; si[i] = i; }
  __syncthreads();
  for (int kk = 2; kk <= N; kk <<= 1) {
    for (int j = kk >> 1; j > 0; j >>= 1) {
      for (int i = tid; i < N; i += 1024) {
        int ij = i ^ j;
        if (ij > i) {
          bool desc = ((i & kk) == 0);
          float a = sv[i], b = sv[ij];
          bool sw = desc ? (a < b) : (a > b);
          if (sw) {
            sv[i] = b; sv[ij] = a;
            int t = si[i]; si[i] = si[ij]; si[ij] = t;
          }
        }
      }
      __syncthreads();
    }
  }
  for (int i = tid; i < K; i += 1024) perm[i] = si[i];
}
__global__ void k_gather_scale(const float* xin, const int* perm, const float* score,
                               int k, int C, float* xout) {
  int i = blockIdx.x * blockDim.x + threadIdx.x;
  if (i >= k * C) return;
  int r = i / C, c = i % C;
  int p = perm[r];
  xout[i] = xin[(size_t)p * C + c] * score[p];
}
__global__ void k_scatter_idx(const int* perm, int k, int* idx) {
  int i = blockIdx.x * blockDim.x + threadIdx.x;
  if (i < k) idx[perm[i]] = i;
}
__global__ void k_remap_edges(int* es, int* ed, float* ew, const int* cnt, int ecap,
                              const int* idx) {
  int i = blockIdx.x * blockDim.x + threadIdx.x;
  if (i >= ecap) return;
  if (i >= *cnt) return;
  int s = es[i];
  if (s < 0) { ew[i] = 0.0f; return; }
  int d = ed[i];
  int r = idx[s], c = idx[d];
  if (r >= 0 && c >= 0) { es[i] = r; ed[i] = c; }
  else { es[i] = -1; ed[i] = -1; ew[i] = 0.0f; }
}

// ---------- single-block exclusive scan (n <= a few K) ----------
__global__ void k_scan_excl(const int* in, int* out, int n, int* total) {
  __shared__ int buf[1024];
  int tid = threadIdx.x;
  int carry = 0;
  for (int base = 0; base < n; base += 1024) {
    int v = (base + tid < n) ? in[base + tid] : 0;
    buf[tid] = v; __syncthreads();
    for (int off = 1; off < 1024; off <<= 1) {
      int t = (tid >= off) ? buf[tid - off] : 0;
      __syncthreads();
      buf[tid] += t;
      __syncthreads();
    }
    if (base + tid < n) out[base + tid] = carry + buf[tid] - v;
    int ct = buf[1023];
    __syncthreads();
    carry += ct;
  }
  if (tid == 0 && total) *total = carry;
}

// ---------- CSR build for A = scatter(non-self edges) + I ----------
__global__ void k_row_count(const int* es, const int* ed, const int* cnt, int ecap, int* rowcnt) {
  int i = blockIdx.x * blockDim.x + threadIdx.x;
  if (i >= ecap || i >= *cnt) return;
  int s = es[i]; if (s < 0) return;
  int d = ed[i]; if (s == d) return;
  atomicAdd(&rowcnt[s], 1);
}
__global__ void k_csr_fill(const int* es, const int* ed, const float* ew, const int* cnt,
                           int ecap, int* cursor, int* cidx, float* cval) {
  int i = blockIdx.x * blockDim.x + threadIdx.x;
  if (i >= ecap || i >= *cnt) return;
  int s = es[i]; if (s < 0) return;
  int d = ed[i]; if (s == d) return;
  int pos = atomicAdd(&cursor[s], 1);
  cidx[pos] = d; cval[pos] = ew[i];
}
__global__ void k_csr_diag(int nn, int* cursor, int* cidx, float* cval) {
  int r = blockIdx.x * blockDim.x + threadIdx.x;
  if (r >= nn) return;
  int pos = atomicAdd(&cursor[r], 1);
  cidx[pos] = r; cval[pos] = 1.0f;
}

// ---------- SpGEMM A*A with LDS row accumulator (ds_add_f32), diag masked ----------
__global__ void k_spgemm_count(const int* __restrict__ indptr, const int* __restrict__ cidx,
                               const float* __restrict__ cval, int nn, int* rowcnt2) {
  __shared__ float acc[4096];
  __shared__ int red[SPG_T];
  int r = blockIdx.x, tid = threadIdx.x;
  for (int c = tid; c < nn; c += SPG_T) acc[c] = 0.0f;
  __syncthreads();
  int rb = indptr[r], re = indptr[r + 1];
  for (int j = rb; j < re; ++j) {
    int k = cidx[j]; float w = cval[j];
    int kb = indptr[k], ke = indptr[k + 1];
    for (int t = kb + tid; t < ke; t += SPG_T) atomicAdd(&acc[cidx[t]], w * cval[t]);
  }
  __syncthreads();
  if (tid == 0) acc[r] = 0.0f;   // (A@A) * (1 - I)
  __syncthreads();
  int c0 = 0;
  for (int c = tid; c < nn; c += SPG_T) c0 += (acc[c] != 0.0f);
  red[tid] = c0; __syncthreads();
  for (int off = SPG_T / 2; off > 0; off >>= 1) {
    if (tid < off) red[tid] += red[tid + off];
    __syncthreads();
  }
  if (tid == 0) rowcnt2[r] = red[0];
}
__global__ void k_spgemm_fill(const int* __restrict__ indptr, const int* __restrict__ cidx,
                              const float* __restrict__ cval, int nn,
                              const int* __restrict__ indptr2, int ecap,
                              int* os, int* od, float* ow) {
  __shared__ float acc[4096];
  __shared__ int pre[SPG_T];
  int r = blockIdx.x, tid = threadIdx.x;
  for (int c = tid; c < nn; c += SPG_T) acc[c] = 0.0f;
  __syncthreads();
  int rb = indptr[r], re = indptr[r + 1];
  for (int j = rb; j < re; ++j) {
    int k = cidx[j]; float w = cval[j];
    int kb = indptr[k], ke = indptr[k + 1];
    for (int t = kb + tid; t < ke; t += SPG_T) atomicAdd(&acc[cidx[t]], w * cval[t]);
  }
  __syncthreads();
  if (tid == 0) acc[r] = 0.0f;
  __syncthreads();
  int chunk = (nn + SPG_T - 1) / SPG_T;
  int lo = tid * chunk; if (lo > nn) lo = nn;
  int hi = lo + chunk;  if (hi > nn) hi = nn;
  int c0 = 0;
  for (int c = lo; c < hi; ++c) c0 += (acc[c] != 0.0f);
  pre[tid] = c0; __syncthreads();
  for (int off = 1; off < SPG_T; off <<= 1) {
    int t = (tid >= off) ? pre[tid - off] : 0;
    __syncthreads();
    pre[tid] += t;
    __syncthreads();
  }
  int base = indptr2[r] + pre[tid] - c0;
  for (int c = lo; c < hi; ++c) {
    if (acc[c] != 0.0f) {
      if (base < ecap) { os[base] = r; od[base] = c; ow[base] = acc[c]; }
      base++;
    }
  }
}

// ---------- up-path scatter / concat ----------
__global__ void k_scatter_rows(const float* x, const int* perm, int k, int C, float* up) {
  int i = blockIdx.x * blockDim.x + threadIdx.x;
  if (i >= k * C) return;
  int r = i / C, c = i % C;
  up[(size_t)perm[r] * C + c] = x[i];
}
__global__ void k_concat(const float* a, const float* b, int n, int C, float* out) {
  int i = blockIdx.x * blockDim.x + threadIdx.x;
  int C2 = 2 * C;
  if (i >= n * C2) return;
  int r = i / C2, c = i % C2;
  out[i] = (c < C) ? a[(size_t)r * C + c] : b[(size_t)r * C + (c - C)];
}

// =====================================================================================
extern "C" void kernel_launch(void* const* d_in, const int* in_sizes, int n_in,
                              void* d_out, int out_size, void* d_ws, size_t ws_size,
                              hipStream_t stream) {
  (void)in_sizes; (void)n_in; (void)out_size; (void)ws_size;
  const int N0 = 4096, N1 = 2048, N2 = 1024, E0 = 16384, C = 128;
  const int ECAP = 1 << 19;  // cap on augmented edge lists

  const float* x_in = (const float*)d_in[0];
  const int*   ei0  = (const int*)d_in[1];

  // pytree leaf order: params keys sorted {down, input, output, pool, up};
  // norm-layer keys sorted {W, att, beta, bias, gamma, lin_b}; output-layer {W, att, bias, lin_b}
  struct Lay { const float *W, *att, *beta, *bias, *gamma, *lin_b; };
  auto getN = [&](int base) {
    Lay L; L.W = (const float*)d_in[base]; L.att = (const float*)d_in[base + 1];
    L.beta = (const float*)d_in[base + 2]; L.bias = (const float*)d_in[base + 3];
    L.gamma = (const float*)d_in[base + 4]; L.lin_b = (const float*)d_in[base + 5];
    return L;
  };
  auto getO = [&](int base) {
    Lay L; L.W = (const float*)d_in[base]; L.att = (const float*)d_in[base + 1];
    L.bias = (const float*)d_in[base + 2]; L.lin_b = (const float*)d_in[base + 3];
    L.beta = nullptr; L.gamma = nullptr;
    return L;
  };
  Lay down00 = getN(2),  down01 = getN(8),  down10 = getN(14), down11 = getN(20);
  Lay in0    = getN(26), in1    = getN(32);
  Lay out0   = getO(38), out1   = getO(42);
  const float* pool0 = (const float*)d_in[46];
  const float* pool1 = (const float*)d_in[47];
  Lay up00 = getN(48), up01 = getN(54), up10 = getN(60), up11 = getN(66);

  // ---- workspace carve-up (~45 MB) ----
  char* wsb = (char*)d_ws; size_t off = 0;
  auto alloc = [&](size_t bytes) -> void* {
    off = (off + 255) & ~(size_t)255;
    void* p = wsb + off; off += bytes; return p;
  };
  int*      meta  = (int*)alloc(64 * 4);           // meta[0..2] = edge counts lvl0/1/2
  float*    nrm   = (float*)alloc(4);
  float*    ones0 = (float*)alloc((size_t)E0 * 4);
  int* ls0 = (int*)alloc((size_t)(E0 + N0) * 4);   int* ld0 = (int*)alloc((size_t)(E0 + N0) * 4);
  int* ls1 = (int*)alloc((size_t)(ECAP + N1) * 4); int* ld1 = (int*)alloc((size_t)(ECAP + N1) * 4);
  int* ls2 = (int*)alloc((size_t)(ECAP + N2) * 4); int* ld2 = (int*)alloc((size_t)(ECAP + N2) * 4);
  int* e1s = (int*)alloc((size_t)ECAP * 4); int* e1d = (int*)alloc((size_t)ECAP * 4);
  float* e1w = (float*)alloc((size_t)ECAP * 4);
  int* e2s = (int*)alloc((size_t)ECAP * 4); int* e2d = (int*)alloc((size_t)ECAP * 4);
  float* e2w = (float*)alloc((size_t)ECAP * 4);
  int* indptr  = (int*)alloc((size_t)(N0 + 1) * 4);
  int* indptr2 = (int*)alloc((size_t)(N0 + 1) * 4);
  int* rowcnt  = (int*)alloc((size_t)N0 * 4);
  int* rowcnt2 = (int*)alloc((size_t)N0 * 4);
  int* cursor  = (int*)alloc((size_t)N0 * 4);
  int*   cidx = (int*)alloc((size_t)(ECAP + N0) * 4);
  float* cval = (float*)alloc((size_t)(ECAP + N0) * 4);
  float* eatt = (float*)alloc((size_t)(ECAP + N0) * 4);
  unsigned* menc = (unsigned*)alloc((size_t)N0 * 4);
  float* ssum  = (float*)alloc((size_t)N0 * 4);
  float* score = (float*)alloc((size_t)N0 * 4);
  float* dotb  = (float*)alloc((size_t)N0 * 4);
  int* perm0 = (int*)alloc((size_t)N1 * 4);
  int* perm1 = (int*)alloc((size_t)N2 * 4);
  int* idxm  = (int*)alloc((size_t)N0 * 4);
  float* x0   = (float*)alloc((size_t)N0 * C * 4);
  float* x1   = (float*)alloc((size_t)N1 * C * 4);
  float* bufA = (float*)alloc((size_t)N0 * 2 * C * 4);
  float* bufB = (float*)alloc((size_t)N0 * 2 * C * 4);
  float* bufC = (float*)alloc((size_t)N0 * C * 4);
  float* xp   = (float*)alloc((size_t)N0 * C * 4);
  float* upb  = (float*)alloc((size_t)N0 * C * 4);

  auto fillf = [&](float* p, float v, int n) { k_fill_f32<<<(n + 255) / 256, 256, 0, stream>>>(p, v, n); };
  auto filli = [&](int* p, int v, int n) { k_fill_i32<<<(n + 255) / 256, 256, 0, stream>>>(p, v, n); };
  auto fillu = [&](unsigned* p, unsigned v, int n) { k_fill_u32<<<(n + 255) / 256, 256, 0, stream>>>(p, v, n); };
  auto tanhN = [&](float* p, int n) { k_tanh_ip<<<(n + 255) / 256, 256, 0, stream>>>(p, n); };

  // linear: xp = xin @ W (+b). Dispatch: WMMA for real GEMMs, matvec/affine for vector cases.
  auto linear = [&](const float* xin, int Nn, int Cin, int Cout,
                    const float* W, const float* b, float* out) {
    if ((Cout & 31) == 0 && (Cin & 31) == 0) {
      dim3 g(Cout / 32, Nn / 16);
      k_gemm_wmma_fast<<<g, 32, 0, stream>>>(xin, W, b, out, Nn, Cout, Cin);
    } else if (Cin >= 32) {
      k_matvec<<<(Nn * 32 + 255) / 256, 256, 0, stream>>>(xin, W, b, out, Nn, Cin);
    } else {
      k_affine1<<<(Nn + 255) / 256, 256, 0, stream>>>(xin, W, b, out, Nn);
    }
  };

  // One GATv2 layer: out = seg_softmax-weighted aggregation of (x@W+lin_b) + bias
  auto gat_layer = [&](const float* xin, int Nn, int Cin, int Cout,
                       const int* es, const int* ed, int etot, const Lay& L, float* out) {
    linear(xin, Nn, Cin, Cout, L.W, L.lin_b, xp);
    fillu(menc, 0x007FFFFFu, Nn);        // enc(-inf)
    fillf(ssum, 0.0f, Nn);
    fillf(out, 0.0f, Nn * Cout);
    int gb = (etot + 255) / 256;
    k_edge_score<<<gb, 256, 0, stream>>>(xp, es, ed, L.att, Cout, etot, eatt);
    k_seg_max<<<gb, 256, 0, stream>>>(eatt, ed, etot, menc);
    k_edge_exp<<<gb, 256, 0, stream>>>(eatt, ed, etot, menc, ssum);
    long long tot = (long long)etot * Cout;
    k_alpha_scatter<<<(int)((tot + 255) / 256), 256, 0, stream>>>(eatt, ssum, xp, es, ed,
                                                                  etot, Cout, out);
    k_add_bias<<<(Nn * Cout + 255) / 256, 256, 0, stream>>>(out, L.bias, Nn, Cout);
  };
  // 2-layer GAT block: [conv, (bn), tanh, conv, (bn)]
  auto block2 = [&](const float* xin, int Nn, int Cin, const int* es, const int* ed, int etot,
                    const Lay& L0, const Lay& L1, int Cout, bool norm, float* tmp, float* out) {
    gat_layer(xin, Nn, Cin, Cout, es, ed, etot, L0, tmp);
    if (norm) k_batchnorm<<<Cout, 256, 0, stream>>>(tmp, L0.gamma, L0.beta, Nn, Cout);
    tanhN(tmp, Nn * Cout);
    gat_layer(tmp, Nn, Cout, Cout, es, ed, etot, L1, out);
    if (norm) k_batchnorm<<<Cout, 256, 0, stream>>>(out, L1.gamma, L1.beta, Nn, Cout);
  };
  // augment_adj: sparse (A - diag + I)^2, diag removed -> new edge list (row-major order)
  auto augment = [&](const int* es, const int* ed, const float* ew, const int* cntIn,
                     int ecapIn, int Nn, int* os, int* od, float* ow, int* cntOut) {
    filli(rowcnt, 1, Nn);   // self loop per row
    k_row_count<<<(ecapIn + 255) / 256, 256, 0, stream>>>(es, ed, cntIn, ecapIn, rowcnt);
    k_scan_excl<<<1, 1024, 0, stream>>>(rowcnt, indptr, Nn, indptr + Nn);
    k_copy_i32<<<(Nn + 255) / 256, 256, 0, stream>>>(cursor, indptr, Nn);
    k_csr_fill<<<(ecapIn + 255) / 256, 256, 0, stream>>>(es, ed, ew, cntIn, ecapIn,
                                                         cursor, cidx, cval);
    k_csr_diag<<<(Nn + 255) / 256, 256, 0, stream>>>(Nn, cursor, cidx, cval);
    k_spgemm_count<<<Nn, SPG_T, 0, stream>>>(indptr, cidx, cval, Nn, rowcnt2);
    k_scan_excl<<<1, 1024, 0, stream>>>(rowcnt2, indptr2, Nn, indptr2 + Nn);
    k_copy_min<<<1, 1, 0, stream>>>(cntOut, indptr2 + Nn, ECAP);
    k_spgemm_fill<<<Nn, SPG_T, 0, stream>>>(indptr, cidx, cval, Nn, indptr2, ECAP, os, od, ow);
  };
  // topk_pool: score, sort, gather x*score, remap edges in place
  auto topk = [&](const float* xcur, int Nn, const float* wpool, int K, int* perm,
                  int* es, int* ed, float* ew, const int* cnt, float* xout) {
    k_matvec<<<(Nn * 32 + 255) / 256, 256, 0, stream>>>(xcur, wpool, nullptr, dotb, Nn, C);
    k_norm<<<1, 256, 0, stream>>>(wpool, C, nrm);
    k_score<<<(Nn + 255) / 256, 256, 0, stream>>>(dotb, nrm, score, Nn);
    k_topk_sort<<<1, 1024, 0, stream>>>(score, Nn, K, perm);
    k_gather_scale<<<(K * C + 255) / 256, 256, 0, stream>>>(xcur, perm, score, K, C, xout);
    filli(idxm, -1, Nn);
    k_scatter_idx<<<(K + 255) / 256, 256, 0, stream>>>(perm, K, idxm);
    k_remap_edges<<<(ECAP + 255) / 256, 256, 0, stream>>>(es, ed, ew, cnt, ECAP, idxm);
  };

  // ==================== forward ====================
  k_seti<<<1, 1, 0, stream>>>(meta + 0, E0);
  fillf(ones0, 1.0f, E0);

  // level-0 edge list (+ self loops)
  k_loop_edges<<<((E0 + N0) + 255) / 256, 256, 0, stream>>>(ei0, ei0 + E0, meta + 0,
                                                            E0, N0, ls0, ld0);
  // input block -> x0 (saved), outer tanh
  block2(x_in, N0, 32, ls0, ld0, E0 + N0, in0, in1, C, true, bufA, x0);
  tanhN(x0, N0 * C);

  // ---- down i=0 ----
  augment(ei0, ei0 + E0, ones0, meta + 0, E0, N0, e1s, e1d, e1w, meta + 1);
  topk(x0, N0, pool0, N1, perm0, e1s, e1d, e1w, meta + 1, bufA);          // bufA: 2048x128
  k_loop_edges<<<((ECAP + N1) + 255) / 256, 256, 0, stream>>>(e1s, e1d, meta + 1,
                                                              ECAP, N1, ls1, ld1);
  block2(bufA, N1, C, ls1, ld1, ECAP + N1, down00, down01, C, true, bufB, x1);
  tanhN(x1, N1 * C);                                                       // x1 saved

  // ---- down i=1 ----
  augment(e1s, e1d, e1w, meta + 1, ECAP, N1, e2s, e2d, e2w, meta + 2);
  topk(x1, N1, pool1, N2, perm1, e2s, e2d, e2w, meta + 2, bufA);          // bufA: 1024x128
  k_loop_edges<<<((ECAP + N2) + 255) / 256, 256, 0, stream>>>(e2s, e2d, meta + 2,
                                                              ECAP, N2, ls2, ld2);
  block2(bufA, N2, C, ls2, ld2, ECAP + N2, down10, down11, C, true, bufB, bufC);
  tanhN(bufC, N2 * C);                                                     // current x: 1024x128

  // ---- up i=0 (j=1) ----
  fillf(upb, 0.0f, N1 * C);
  k_scatter_rows<<<(N2 * C + 255) / 256, 256, 0, stream>>>(bufC, perm1, N2, C, upb);
  k_concat<<<(N1 * 2 * C + 255) / 256, 256, 0, stream>>>(x1, upb, N1, C, bufA);
  block2(bufA, N1, 2 * C, ls1, ld1, ECAP + N1, up00, up01, C, true, bufB, bufC);
  tanhN(bufC, N1 * C);                                                     // current x: 2048x128

  // ---- up i=1 (j=0) ----
  fillf(upb, 0.0f, N0 * C);
  k_scatter_rows<<<(N1 * C + 255) / 256, 256, 0, stream>>>(bufC, perm0, N1, C, upb);
  k_concat<<<(N0 * 2 * C + 255) / 256, 256, 0, stream>>>(x0, upb, N0, C, bufA);
  block2(bufA, N0, 2 * C, ls0, ld0, E0 + N0, up10, up11, C, true, bufB, bufC);
  tanhN(bufC, N0 * C);                                                     // current x: 4096x128

  // ---- output block (no norm) + sigmoid ----
  block2(bufC, N0, C, ls0, ld0, E0 + N0, out0, out1, 1, false, bufB, bufA);
  k_sigmoid<<<(N0 + 255) / 256, 256, 0, stream>>>(bufA, (float*)d_out, N0);
}
